// HINN_10711648436928
// MI455X (gfx1250) — matrix operand order repaired and checked by reading, not compiled, not run on hardware
//
#include <hip/hip_runtime.h>
#include <hip/hip_bf16.h>

typedef __attribute__((ext_vector_type(16))) _Float16 v16h;
typedef __attribute__((ext_vector_type(8)))  _Float16 v8h;
typedef __attribute__((ext_vector_type(8)))  float    v8f;

union F16x16 { v16h v; v8h h[2]; };

// ---------------------------------------------------------------------------
// Small utility: zero a float region (stats / border accumulators).
// ---------------------------------------------------------------------------
__global__ void zero_kernel(float* p, int n) {
    int i = blockIdx.x * blockDim.x + threadIdx.x;
    if (i < n) p[i] = 0.f;
}

// ---------------------------------------------------------------------------
// Weight repack: f32 [cout][cin][3][3]  ->  f16 [cout][Kpad] with
// K ordered (r,c,cin) and zero-padded to Kpad (multiple of 32).
// This makes every WMMA A-fragment two aligned 16B loads, no bounds checks.
// ---------------------------------------------------------------------------
__global__ void wcvt_kernel(const float* __restrict__ src, _Float16* __restrict__ dst,
                            int Cout, int Cin, int Kpad) {
    int i = blockIdx.x * blockDim.x + threadIdx.x;
    int total = Cout * Kpad;
    if (i >= total) return;
    int co = i / Kpad, k = i % Kpad;
    float v = 0.f;
    if (k < Cin * 9) {
        int rc = k / Cin, cin = k % Cin;           // K = (r*3+c)*Cin + cin
        v = src[(co * Cin + cin) * 9 + rc];
    }
    dst[i] = (_Float16)v;
}

// ---------------------------------------------------------------------------
// him(): bilinear downsample (align_corners=True) 256->128, fused with
// per-sample max|v| reduction (border term) via bit-pattern atomicMax.
// ---------------------------------------------------------------------------
__global__ void downsample_kernel(const float* __restrict__ x, float* __restrict__ ds,
                                  unsigned* __restrict__ bord, int Bn) {
    const int total = Bn * 2 * 128 * 128;
    int i = blockIdx.x * blockDim.x + threadIdx.x;
    if (i >= total) return;
    int ox = i & 127, oy = (i >> 7) & 127;
    int c  = (i >> 14) & 1, b = i >> 15;
    const float step = 255.f / 127.f;
    float sy = oy * step, sx = ox * step;
    int y0 = (int)floorf(sy), x0 = (int)floorf(sx);
    int y1 = min(y0 + 1, 255), x1 = min(x0 + 1, 255);
    float wy = sy - y0, wx = sx - x0;
    const float* p = x + ((size_t)(b * 2 + c)) * 65536;
    float v = p[y0 * 256 + x0] * (1.f - wy) * (1.f - wx)
            + p[y1 * 256 + x0] * wy * (1.f - wx)
            + p[y0 * 256 + x1] * (1.f - wy) * wx
            + p[y1 * 256 + x1] * wy * wx;
    ds[i] = v;
    atomicMax(&bord[b], __float_as_uint(fabsf(v)));
}

// ---------------------------------------------------------------------------
// edge = crowd * (1 - erode3x3(crowd)),  crowd = (|vel| > 0.2)
// ---------------------------------------------------------------------------
__global__ void edge_kernel(const float* __restrict__ ds, float* __restrict__ edge, int Bn) {
    const int total = Bn * 128 * 128;
    int i = blockIdx.x * blockDim.x + threadIdx.x;
    if (i >= total) return;
    int x = i & 127, y = (i >> 7) & 127, b = i >> 14;
    const float* u = ds + (size_t)b * 2 * 16384;
    const float* v = u + 16384;
    auto crowd = [&](int yy, int xx) -> float {
        float uu = u[yy * 128 + xx], vv = v[yy * 128 + xx];
        return (sqrtf(uu * uu + vv * vv) > 0.2f) ? 1.f : 0.f;
    };
    float c0 = crowd(y, x);
    float er = 1.f;
    for (int dy = -1; dy <= 1; ++dy)
        for (int dx = -1; dx <= 1; ++dx) {
            int yy = y + dy, xx = x + dx;
            if (yy >= 0 && yy < 128 && xx >= 0 && xx < 128)
                er = fminf(er, crowd(yy, xx));
        }
    edge[i] = c0 * (1.f - er);
}

// cohesion = dilate5x5(edge)
__global__ void cohesion_kernel(const float* __restrict__ edge, float* __restrict__ coh, int Bn) {
    const int total = Bn * 128 * 128;
    int i = blockIdx.x * blockDim.x + threadIdx.x;
    if (i >= total) return;
    int x = i & 127, y = (i >> 7) & 127, b = i >> 14;
    const float* e = edge + (size_t)b * 16384;
    float m = 0.f;
    for (int dy = -2; dy <= 2; ++dy)
        for (int dx = -2; dx <= 2; ++dx) {
            int yy = y + dy, xx = x + dx;
            if (yy >= 0 && yy < 128 && xx >= 0 && xx < 128)
                m = fmaxf(m, e[yy * 128 + xx]);
        }
    coh[i] = m;
}

// ---------------------------------------------------------------------------
// navier_stokes PDE body (stencils + sign-gathers + cohesion term).
// ---------------------------------------------------------------------------
__global__ void navier_kernel(const float* __restrict__ ds, const float* __restrict__ coh,
                              const unsigned* __restrict__ bord, float* __restrict__ pde, int Bn) {
    const int total = Bn * 128 * 128;
    int i = blockIdx.x * blockDim.x + threadIdx.x;
    if (i >= total) return;
    int x = i & 127, y = (i >> 7) & 127, b = i >> 14;
    const float* u = ds + (size_t)b * 2 * 16384;
    const float* v = u + 16384;

    float un[3][3], vn[3][3];
    for (int r = 0; r < 3; ++r)
        for (int c = 0; c < 3; ++c) {
            int yy = y + r - 1, xx = x + c - 1;
            bool ok = (yy >= 0 && yy < 128 && xx >= 0 && xx < 128);
            un[r][c] = ok ? u[yy * 128 + xx] : 0.f;
            vn[r][c] = ok ? v[yy * 128 + xx] : 0.f;
        }
    float uc = un[1][1], vc = vn[1][1];

    float u_x = -un[0][0] + un[0][2] - 2.f * un[1][0] + 2.f * un[1][2] - un[2][0] + un[2][2];
    float v_x = -vn[0][0] + vn[0][2] - 2.f * vn[1][0] + 2.f * vn[1][2] - vn[2][0] + vn[2][2];
    float u_y = -un[0][0] - 2.f * un[0][1] - un[0][2] + un[2][0] + 2.f * un[2][1] + un[2][2];
    float v_y = -vn[0][0] - 2.f * vn[0][1] - vn[0][2] + vn[2][0] + 2.f * vn[2][1] + vn[2][2];
    float u_lap = un[0][1] + un[1][0] + un[1][2] + un[2][1] - 4.f * uc;
    float v_lap = vn[0][1] + vn[1][0] + vn[1][2] + vn[2][1] - 4.f * vc;
    float u_avg = 0.125f * (un[0][0] + un[0][1] + un[0][2] + un[1][0] + un[1][2] + un[2][0] + un[2][1] + un[2][2]);
    float v_avg = 0.125f * (vn[0][0] + vn[0][1] + vn[0][2] + vn[1][0] + vn[1][2] + vn[2][0] + vn[2][1] + vn[2][2]);

    float us = (uc > 0.f) ? 1.f : ((uc < 0.f) ? -1.f : 0.f);
    float vs = (vc > 0.f) ? 1.f : ((vc < 0.f) ? -1.f : 0.f);
    int ix = min(127, max(0, x + (int)(us * 10.f)));
    int iy = min(127, max(0, y + (int)(vs * 10.f)));
    float u_nav = u[iy * 128 + ix] - uc;
    float v_nav = v[iy * 128 + ix] - vc;

    auto magAt = [&](int yy, int xx) -> float {
        float uu = u[yy * 128 + xx], vv = v[yy * 128 + xx];
        return sqrtf(uu * uu + vv * vv);
    };
    float side1 = magAt(y, ix);
    float side2 = magAt(iy, x);
    float coh_sign = (side1 > side2) ? 1.f : 0.f;
    float border = __uint_as_float(bord[b]) * 0.4f;
    float ch = coh[(size_t)b * 16384 + y * 128 + x];
    float u_coh = ch * coh_sign * us * border;
    float v_coh = -ch * coh_sign * vs * border;

    const float VISC = 0.2f, KK = 0.5f;
    float u_pde = uc * u_x + vc * u_y - VISC * (u_lap + u_lap) + KK * (u_avg - uc) + u_nav + u_coh;
    float v_pde = uc * v_x + vc * v_y - VISC * (v_lap + v_lap) + KK * (v_avg - vc) + v_nav + v_coh;
    pde[(size_t)(b * 2 + 0) * 16384 + y * 128 + x] = u_pde;
    pde[(size_t)(b * 2 + 1) * 16384 + y * 128 + x] = v_pde;
}

// ---------------------------------------------------------------------------
// bicubic upsample 128->256, a=-0.75, align_corners=False
// ---------------------------------------------------------------------------
__device__ __forceinline__ float cubic_k(float xx) {
    const float a = -0.75f;
    float ax = fabsf(xx);
    if (ax <= 1.f) return (a + 2.f) * ax * ax * ax - (a + 3.f) * ax * ax + 1.f;
    if (ax < 2.f)  return a * ax * ax * ax - 5.f * a * ax * ax + 8.f * a * ax - 4.f * a;
    return 0.f;
}

__global__ void upsample_kernel(const float* __restrict__ pde, float* __restrict__ xnse, int Bn) {
    const int total = Bn * 2 * 256 * 256;
    int i = blockIdx.x * blockDim.x + threadIdx.x;
    if (i >= total) return;
    int ox = i & 255, oy = (i >> 8) & 255;
    int c = (i >> 16) & 1, b = i >> 17;
    float sy = (oy + 0.5f) * 0.5f - 0.5f;
    float sx = (ox + 0.5f) * 0.5f - 0.5f;
    int iy0 = (int)floorf(sy), ix0 = (int)floorf(sx);
    float ty = sy - iy0, tx = sx - ix0;
    float wy[4] = { cubic_k(ty + 1.f), cubic_k(ty), cubic_k(1.f - ty), cubic_k(2.f - ty) };
    float wx[4] = { cubic_k(tx + 1.f), cubic_k(tx), cubic_k(1.f - tx), cubic_k(2.f - tx) };
    const float* p = pde + (size_t)(b * 2 + c) * 16384;
    float acc = 0.f;
    for (int j = 0; j < 4; ++j) {
        int yy = min(127, max(0, iy0 - 1 + j));
        float row = 0.f;
        for (int k = 0; k < 4; ++k) {
            int xx = min(127, max(0, ix0 - 1 + k));
            row += wx[k] * p[yy * 128 + xx];
        }
        acc += wy[j] * row;
    }
    xnse[i] = acc;
}

// ---------------------------------------------------------------------------
// conv1: cat([x, x_nse]) -> 3x3 stride 2, 4->16 ch, tanh.  Cin=4 -> direct VALU.
// ---------------------------------------------------------------------------
__global__ void conv1_kernel(const float* __restrict__ x, const float* __restrict__ xnse,
                             const float* __restrict__ w, const float* __restrict__ bias,
                             float* __restrict__ out, int Bn) {
    const int total = Bn * 16 * 128 * 128;
    int i = blockIdx.x * blockDim.x + threadIdx.x;
    if (i >= total) return;
    int ox = i & 127, oy = (i >> 7) & 127;
    int co = (i >> 14) & 15, b = i >> 18;
    float acc = bias[co];
    for (int ci = 0; ci < 4; ++ci) {
        const float* src = (ci < 2) ? (x + (size_t)(b * 2 + ci) * 65536)
                                    : (xnse + (size_t)(b * 2 + (ci - 2)) * 65536);
        for (int r = 0; r < 3; ++r)
            for (int c = 0; c < 3; ++c) {
                int iy = 2 * oy + r - 1, ixx = 2 * ox + c - 1;
                if (iy >= 0 && iy < 256 && ixx >= 0 && ixx < 256)
                    acc += w[((co * 4 + ci) * 3 + r) * 3 + c] * src[iy * 256 + ixx];
            }
    }
    out[i] = tanhf(acc);
}

// ---------------------------------------------------------------------------
// Implicit-GEMM 3x3 conv (stride 1, pad 1, H=W=128, Cout=32) on WMMA f16->f32.
//
// K ordering is (r, c, cin) so one 32-wide K chunk is channel-contiguous:
//  - A fragment: two aligned global b128 loads from the pre-packed f16 weights
//  - B fragment: two aligned ds_load_b128 from an LDS-staged halo patch
// Block = 128 threads (4 waves); block tile = 64 pixels of one row x 32 couts.
// Each wave: 16 pixels x 32 couts -> 2 accumulators, 2 WMMA per K chunk,
// fully unrolled (Cin templated).
//
// Fragment layouts used by this kernel (wave32, contiguous K per lane half):
//   A (16x32 f16): row M=lane&15, lane half selects K {0..15}/{16..31}
//   B (32x16 f16): lane half selects K {0..15}/{16..31}, column = lane&15
//   D (16x16 f32): VGPR r = row (half_sel*8 + r), lane = column
// ---------------------------------------------------------------------------
template <int CIN>
__global__ __launch_bounds__(128)
void conv3x3_wmma_kernel(const float* __restrict__ in, const _Float16* __restrict__ w16,
                         const float* __restrict__ bias, float* __restrict__ out,
                         int fuse_tanh) {
    constexpr int KTOT = CIN * 9;
    constexpr int KPAD = (KTOT + 31) & ~31;
    constexpr int NCH  = KPAD / 32;
    constexpr int PCOL = 66;                 // 64-pixel tile + 1 halo each side
    __shared__ _Float16 patch[3 * PCOL * CIN];

    const int tid  = threadIdx.x;
    // block -> (b, py, 64-pixel half-row)
    const int blk  = blockIdx.x;
    const int tile = blk & 1;
    const int py   = (blk >> 1) & 127;
    const int b    = blk >> 8;
    const int px_base = tile * 64;

    // ---- cooperative f32 -> f16 halo staging: patch[(row*PCOL + col)*CIN + cin]
    const float* inb = in + (size_t)b * CIN * 16384;
    for (int i = tid; i < 3 * PCOL * CIN; i += 128) {
        int row = i / (PCOL * CIN);
        int rem = i - row * (PCOL * CIN);
        int col = rem / CIN;
        int cin = rem - col * CIN;
        int gy = py + row - 1;
        int gx = px_base + col - 1;
        float v = 0.f;
        if (gy >= 0 && gy < 128 && gx >= 0 && gx < 128)
            v = inb[cin * 16384 + gy * 128 + gx];
        patch[i] = (_Float16)v;
    }
    __syncthreads();

    const int lane     = tid & 31;
    const int half_sel = lane >> 4;
    const int mcol     = lane & 15;          // A row within cout tile / D column pixel
    const int wave     = tid >> 5;
    const int px_local = wave * 16 + mcol;   // 0..63
    const int px       = px_base + px_local;

    // speculative prefetch of the weight rows this lane streams (global_prefetch_b8)
    __builtin_prefetch(w16 + (size_t)mcol * KPAD, 0, 1);

    v8f acc0 = {};
    v8f acc1 = {};
#pragma unroll
    for (int ch = 0; ch < NCH; ++ch) {
        const int kb  = ch * 32;
        const int kbh = kb + half_sel * 16;   // K base for this lane-half
        // ---- A fragments (weights): couts 0-15 and 16-31
        const int ka0 = kb + half_sel * 16;   // halfs j=0..7
        F16x16 a0, a1;
        a0.h[0] = *(const v8h*)(w16 + (size_t)mcol * KPAD + ka0);
        a0.h[1] = *(const v8h*)(w16 + (size_t)mcol * KPAD + ka0 + 8);
        a1.h[0] = *(const v8h*)(w16 + (size_t)(16 + mcol) * KPAD + ka0);
        a1.h[1] = *(const v8h*)(w16 + (size_t)(16 + mcol) * KPAD + ka0 + 8);
        // ---- B fragment (activations from LDS)
        int rc    = kbh / CIN;                // stencil tap for this lane-half
        int cin0  = kbh - rc * CIN;           // first channel of the 16 contiguous
        bool valid = (rc < 9);                // K zero-padding (Cin=16 tail only)
        int rcc = valid ? rc : 0;
        int rr = rcc / 3, cc = rcc - rr * 3;
        int off = ((rr * PCOL) + px_local + cc) * CIN + cin0;
        F16x16 bf;
        bf.h[0] = *(const v8h*)&patch[off];
        bf.h[1] = *(const v8h*)&patch[off + 8];
        if (!valid) {
            v8h z = {};
            bf.h[0] = z;
            bf.h[1] = z;
        }
        acc0 = __builtin_amdgcn_wmma_f32_16x16x32_f16(false, a0.v, false, bf.v,
                                                      (short)0, acc0, false, false);
        acc1 = __builtin_amdgcn_wmma_f32_16x16x32_f16(false, a1.v, false, bf.v,
                                                      (short)0, acc1, false, false);
    }

    // ---- epilogue: D rows are couts, D column is this lane's pixel
    float* outb = out + (size_t)b * 32 * 16384 + py * 128 + px;
#pragma unroll
    for (int r = 0; r < 8; ++r) {
        int co0 = half_sel * 8 + r;
        float v0 = acc0[r] + bias[co0];
        float v1 = acc1[r] + bias[co0 + 16];
        if (fuse_tanh) { v0 = tanhf(v0); v1 = tanhf(v1); }
        outb[(size_t)co0 * 16384] = v0;
        outb[(size_t)(co0 + 16) * 16384] = v1;
    }
}

// ---------------------------------------------------------------------------
// LayerNorm over (C,H,W) per sample: pass 1 partial sums -> f32 atomics.
// ---------------------------------------------------------------------------
__global__ void ln_reduce_kernel(const float* __restrict__ r, float* __restrict__ stats,
                                 int per_sample) {
    __shared__ float s1[256], s2[256];
    int b = blockIdx.y;
    int i = blockIdx.x * blockDim.x + threadIdx.x;
    float v = (i < per_sample) ? r[(size_t)b * per_sample + i] : 0.f;
    s1[threadIdx.x] = v;
    s2[threadIdx.x] = v * v;
    __syncthreads();
    for (int s = 128; s > 0; s >>= 1) {
        if (threadIdx.x < s) {
            s1[threadIdx.x] += s1[threadIdx.x + s];
            s2[threadIdx.x] += s2[threadIdx.x + s];
        }
        __syncthreads();
    }
    if (threadIdx.x == 0) {
        atomicAdd(&stats[2 * b + 0], s1[0]);
        atomicAdd(&stats[2 * b + 1], s2[0]);
    }
}

// pass 2: y += (r - mu)/sqrt(var+eps) * ln_w + ln_b   (residual fused)
__global__ void ln_apply_kernel(float* __restrict__ y, const float* __restrict__ r,
                                const float* __restrict__ lnw, const float* __restrict__ lnb,
                                const float* __restrict__ stats, int per_sample, int total) {
    int i = blockIdx.x * blockDim.x + threadIdx.x;
    if (i >= total) return;
    int b = i / per_sample;
    int j = i % per_sample;
    float inv_n = 1.f / (float)per_sample;
    float mu = stats[2 * b] * inv_n;
    float var = stats[2 * b + 1] * inv_n - mu * mu;
    float rn = (r[i] - mu) * rsqrtf(var + 1e-5f) * lnw[j] + lnb[j];
    y[i] += rn;
}

// ---------------------------------------------------------------------------
// conv3 (2->2, 3x3, stride 2) with pixel_shuffle(r=4) folded into indexing:
// ps[b,c,hh,ww] = y[b, c*16 + (hh%4)*4 + (ww%4), hh/4, ww/4]
// ---------------------------------------------------------------------------
__global__ void conv3_ps_kernel(const float* __restrict__ y, const float* __restrict__ w,
                                const float* __restrict__ bias, float* __restrict__ out, int Bn) {
    const int total = Bn * 2 * 256 * 256;
    int i = blockIdx.x * blockDim.x + threadIdx.x;
    if (i >= total) return;
    int ox = i & 255, oy = (i >> 8) & 255;
    int co = (i >> 16) & 1, b = i >> 17;
    float acc = bias[co];
    for (int ci = 0; ci < 2; ++ci)
        for (int r = 0; r < 3; ++r)
            for (int c = 0; c < 3; ++c) {
                int hh = 2 * oy + r - 1, ww = 2 * ox + c - 1;
                if (hh >= 0 && hh < 512 && ww >= 0 && ww < 512) {
                    int chn = ci * 16 + (hh & 3) * 4 + (ww & 3);
                    acc += w[((co * 2 + ci) * 3 + r) * 3 + c]
                         * y[((size_t)(b * 32 + chn) * 128 + (hh >> 2)) * 128 + (ww >> 2)];
                }
            }
    out[i] = acc;
}

// ---------------------------------------------------------------------------
// Orchestration
// ---------------------------------------------------------------------------
extern "C" void kernel_launch(void* const* d_in, const int* in_sizes, int n_in,
                              void* d_out, int out_size, void* d_ws, size_t ws_size,
                              hipStream_t stream) {
    (void)in_sizes; (void)n_in; (void)out_size; (void)ws_size;
    const float* x       = (const float*)d_in[0];
    const float* conv1_w = (const float*)d_in[1];
    const float* conv1_b = (const float*)d_in[2];
    const float* conv2_w = (const float*)d_in[3];
    const float* conv2_b = (const float*)d_in[4];
    const float* bc1_w   = (const float*)d_in[5];
    const float* bc1_b   = (const float*)d_in[6];
    const float* bc2_w   = (const float*)d_in[7];
    const float* bc2_b   = (const float*)d_in[8];
    const float* ln_w    = (const float*)d_in[9];
    const float* ln_b    = (const float*)d_in[10];
    const float* conv3_w = (const float*)d_in[11];
    const float* conv3_b = (const float*)d_in[12];
    float* out = (float*)d_out;

    constexpr int Bn = 8;
    constexpr int HW = 256 * 256;   // 65536
    constexpr int hw = 128 * 128;   // 16384
    constexpr int CHW = 32 * hw;    // per-sample LN extent
    constexpr int KPAD16 = 160;     // Cin=16 -> Ktot 144 padded
    constexpr int KPAD32 = 288;     // Cin=32 -> Ktot 288 (already mult of 32)

    float* ws   = (float*)d_ws;
    float* ds   = ws;                       // B*2*hw
    float* pde  = ds   + (size_t)Bn * 2 * hw;
    float* edge = pde  + (size_t)Bn * 2 * hw;
    float* coh  = edge + (size_t)Bn * hw;
    float* xnse = coh  + (size_t)Bn * hw;   // B*2*HW
    float* y1   = xnse + (size_t)Bn * 2 * HW;  // B*16*hw
    float* ybuf = y1   + (size_t)Bn * 16 * hw; // B*32*hw
    float* rbuf = ybuf + (size_t)Bn * CHW;
    float* tbuf = rbuf + (size_t)Bn * CHW;
    float* stats = tbuf + (size_t)Bn * CHW;    // 2*Bn floats
    unsigned* bord = (unsigned*)(stats + 2 * Bn); // Bn u32
    _Float16* w16_conv2 = (_Float16*)(bord + Bn);     // 32*KPAD16 (16B aligned)
    _Float16* w16_blk   = w16_conv2 + 32 * KPAD16;    // 8 * 32*KPAD32

    int n;
    // reset border accumulators (+stats region) each call (graph replay safe)
    zero_kernel<<<1, 64, 0, stream>>>(stats, 2 * Bn + Bn);

    // ---- repack all WMMA conv weights to padded f16 (r,c,cin) layout ----
    n = 32 * KPAD16;
    wcvt_kernel<<<(n + 255) / 256, 256, 0, stream>>>(conv2_w, w16_conv2, 32, 16, KPAD16);
    n = 32 * KPAD32;
    for (int i = 0; i < 4; ++i) {
        wcvt_kernel<<<(n + 255) / 256, 256, 0, stream>>>(
            bc1_w + (size_t)i * 32 * 32 * 9, w16_blk + (size_t)(2 * i) * 32 * KPAD32,
            32, 32, KPAD32);
        wcvt_kernel<<<(n + 255) / 256, 256, 0, stream>>>(
            bc2_w + (size_t)i * 32 * 32 * 9, w16_blk + (size_t)(2 * i + 1) * 32 * KPAD32,
            32, 32, KPAD32);
    }

    // ---- him() ----
    n = Bn * 2 * hw;
    downsample_kernel<<<(n + 255) / 256, 256, 0, stream>>>(x, ds, bord, Bn);
    n = Bn * hw;
    edge_kernel<<<(n + 255) / 256, 256, 0, stream>>>(ds, edge, Bn);
    cohesion_kernel<<<(n + 255) / 256, 256, 0, stream>>>(edge, coh, Bn);
    navier_kernel<<<(n + 255) / 256, 256, 0, stream>>>(ds, coh, bord, pde, Bn);
    n = Bn * 2 * HW;
    upsample_kernel<<<(n + 255) / 256, 256, 0, stream>>>(pde, xnse, Bn);

    // ---- encoder ----
    n = Bn * 16 * hw;
    conv1_kernel<<<(n + 255) / 256, 256, 0, stream>>>(x, xnse, conv1_w, conv1_b, y1, Bn);

    const int nblk = Bn * 128 * 2;    // 64-pixel half-rows
    conv3x3_wmma_kernel<16><<<nblk, 128, 0, stream>>>(y1, w16_conv2, conv2_b, ybuf, 1);

    // ---- 4 residual blocks (WMMA convs + global LayerNorm) ----
    for (int i = 0; i < 4; ++i) {
        conv3x3_wmma_kernel<32><<<nblk, 128, 0, stream>>>(
            ybuf, w16_blk + (size_t)(2 * i) * 32 * KPAD32, bc1_b + i * 32, tbuf, 1);
        conv3x3_wmma_kernel<32><<<nblk, 128, 0, stream>>>(
            tbuf, w16_blk + (size_t)(2 * i + 1) * 32 * KPAD32, bc2_b + i * 32, rbuf, 0);
        zero_kernel<<<1, 64, 0, stream>>>(stats, 2 * Bn);
        dim3 gred(CHW / 256, Bn);
        ln_reduce_kernel<<<gred, 256, 0, stream>>>(rbuf, stats, CHW);
        n = Bn * CHW;
        ln_apply_kernel<<<(n + 255) / 256, 256, 0, stream>>>(ybuf, rbuf,
                                                             ln_w + (size_t)i * CHW,
                                                             ln_b + (size_t)i * CHW,
                                                             stats, CHW, n);
    }

    // ---- pixel_shuffle(4) + conv3 stride 2 ----
    n = Bn * 2 * HW;
    conv3_ps_kernel<<<(n + 255) / 256, 256, 0, stream>>>(ybuf, conv3_w, conv3_b, out, Bn);
}
